// GNN_44083544326840
// MI455X (gfx1250) — compile-verified, matrix operand order checked by reading, not run on hardware
//
#include <hip/hip_runtime.h>
#include <hip/hip_bf16.h>

// CDNA5 (gfx1250) WMMA types, per probe-verified builtin signatures.
typedef __attribute__((ext_vector_type(16))) __bf16 v16bf;
typedef __attribute__((ext_vector_type(8)))  float  v8f;

#define N_NODES 100000
#define N_EDGES 1600000

// ---------------------------------------------------------------------------
// Utility: zero a float region (agg buffers / degree counters each call).
// ---------------------------------------------------------------------------
__global__ void zero_f32(float* __restrict__ p, long n) {
    long i = (long)blockIdx.x * blockDim.x + threadIdx.x;
    if (i < n) p[i] = 0.0f;
}

// ---------------------------------------------------------------------------
// Degrees via float atomics (segment_sum of ones).
// ---------------------------------------------------------------------------
__global__ void degree_kernel(const int* __restrict__ src, const int* __restrict__ dst,
                              float* __restrict__ outdeg, float* __restrict__ indeg, int E) {
    int e = blockIdx.x * blockDim.x + threadIdx.x;
    if (e < E) {
        atomicAdd(&outdeg[src[e]], 1.0f);
        atomicAdd(&indeg[dst[e]], 1.0f);
    }
}

__global__ void norm_kernel(const float* __restrict__ outdeg, const float* __restrict__ indeg,
                            float* __restrict__ outnorm, float* __restrict__ innorm, int n) {
    int i = blockIdx.x * blockDim.x + threadIdx.x;
    if (i < n) {
        outnorm[i] = rsqrtf(fmaxf(outdeg[i], 1.0f));
        innorm[i]  = rsqrtf(fmaxf(indeg[i], 1.0f));
    }
}

// ---------------------------------------------------------------------------
// Pre-pack weight matrix W [K x NCOLS] (row-major f32) into bf16 B-fragments
// in the 16-bit 32x16 WMMA B layout: lane holds column (lane&15); K packed as
// element j -> K = 32*kstep + 8*(lane>>4) + (j<8 ? j : j+8).
// One 32-thread block per (kstep, coltile) fragment.
// ---------------------------------------------------------------------------
template<int KSTEPS, int NCT>
__global__ void prep_bfrag(const float* __restrict__ W, v16bf* __restrict__ frag, int ncols) {
    int b    = blockIdx.x;            // 0 .. KSTEPS*NCT-1
    int ks   = b / NCT;
    int ct   = b % NCT;
    int lane = threadIdx.x & 31;
    int col  = ct * 16 + (lane & 15);
    int kb   = ks * 32 + 8 * (lane >> 4);
    v16bf v;
#pragma unroll
    for (int j = 0; j < 16; ++j) {
        int k = kb + (j < 8 ? j : j + 8);
        v[j] = (__bf16)W[k * ncols + col];
    }
    frag[b * 32 + lane] = v;
}

// ---------------------------------------------------------------------------
// H[N, NCT*16] = (X[N, KSTEPS*32] * norm[:,None]) @ W   via v_wmma_f32_16x16x32_bf16.
// One wave per 16-row tile; NCT 16x16 accumulator tiles across columns.
// N_NODES == 6250 * 16 exactly, so no row remainder; waves exit uniformly
// (EXEC all-1s at every WMMA, as the ISA requires).
// ---------------------------------------------------------------------------
template<int KSTEPS, int NCT>
__global__ void gemm_wmma(const float* __restrict__ X, const float* __restrict__ norm,
                          const v16bf* __restrict__ bfrag, float* __restrict__ H) {
    const int K    = KSTEPS * 32;
    const int NOUT = NCT * 16;
    const int ntiles = N_NODES / 16;

    int wave = (int)((blockIdx.x * blockDim.x + threadIdx.x) >> 5);
    int lane = threadIdx.x & 31;
    if (wave >= ntiles) return;               // wave-uniform exit

    int row   = wave * 16 + (lane & 15);
    float s   = norm[row];
    const float* xr = X + (long)row * K + 8 * (lane >> 4);

    v8f acc[NCT] = {};

#pragma unroll
    for (int ks = 0; ks < KSTEPS; ++ks) {
        // A fragment: two contiguous 32B reads of this row, scaled, cvt to bf16.
        float4 f[4];
        const float4* p0 = (const float4*)(xr + ks * 32);
        const float4* p1 = (const float4*)(xr + ks * 32 + 16);
        f[0] = p0[0]; f[1] = p0[1]; f[2] = p1[0]; f[3] = p1[1];
        const float* tf = (const float*)f;
        v16bf a;
#pragma unroll
        for (int j = 0; j < 16; ++j) a[j] = (__bf16)(tf[j] * s);

#pragma unroll
        for (int ct = 0; ct < NCT; ++ct) {
            v16bf bb = bfrag[(ks * NCT + ct) * 32 + lane];
            acc[ct] = __builtin_amdgcn_wmma_f32_16x16x32_bf16(
                false, a, false, bb, (short)0, acc[ct], false, false);
        }
    }

    // C/D layout: element r of lane l is D[M = r + 8*(l>>4)][N = l&15].
    int rbase = wave * 16 + 8 * (lane >> 4);
    int col0  = lane & 15;
#pragma unroll
    for (int ct = 0; ct < NCT; ++ct) {
#pragma unroll
        for (int r = 0; r < 8; ++r) {
            H[(long)(rbase + r) * NOUT + ct * 16 + col0] = acc[ct][r];
        }
    }
}

// ---------------------------------------------------------------------------
// Edge gather + scatter-add: thread per (edge, 4-float chunk).
// float4 coalesced gather of h[src], 4x global_atomic_add_f32 into agg[dst].
// ---------------------------------------------------------------------------
template<int F>
__global__ void scatter_add(const float* __restrict__ H, const int* __restrict__ src,
                            const int* __restrict__ dst, float* __restrict__ agg) {
    const int C = F / 4;
    long t = (long)blockIdx.x * blockDim.x + threadIdx.x;
    if (t >= (long)N_EDGES * C) return;
    int e = (int)(t / C);
    int c = (int)(t % C);
    int s = src[e];
    int d = dst[e];
    float4 v = *(const float4*)(H + (long)s * F + c * 4);
    float* o = agg + (long)d * F + c * 4;
    atomicAdd(o + 0, v.x);
    atomicAdd(o + 1, v.y);
    atomicAdd(o + 2, v.z);
    atomicAdd(o + 3, v.w);
}

// ---------------------------------------------------------------------------
// Epilogue: out = agg * in_norm[row] + bias[col], optional relu.
// ---------------------------------------------------------------------------
template<int F, bool RELU>
__global__ void epilogue_kernel(const float* __restrict__ agg, const float* __restrict__ innorm,
                                const float* __restrict__ bias, float* __restrict__ out) {
    long t = (long)blockIdx.x * blockDim.x + threadIdx.x;
    if (t >= (long)N_NODES * F) return;
    int row = (int)(t / F);
    int col = (int)(t % F);
    float v = agg[t] * innorm[row] + bias[col];
    if (RELU) v = fmaxf(v, 0.0f);
    out[t] = v;
}

// ---------------------------------------------------------------------------
// Launch. Workspace layout (floats, N = 100000):
//   [0,N)      outdeg        [N,2N)    indeg
//   [2N,3N)    outnorm       [3N,4N)   innorm
//   [4N,68N)   h1 (layer-1 GEMM out); after layer-1 scatter this region is
//              recycled: h2 @ [4N,36N), agg2 @ [36N,68N)
//   [68N,132N) agg1 -> y1 (relu epilogue in place)
//   @132N      bf16 weight fragments (layer1: 16 frags, layer2: 4 frags)
// Total ~52.8 MB + 20 KB.
// ---------------------------------------------------------------------------
extern "C" void kernel_launch(void* const* d_in, const int* in_sizes, int n_in,
                              void* d_out, int out_size, void* d_ws, size_t ws_size,
                              hipStream_t stream) {
    const float* X   = (const float*)d_in[0];
    const int*   src = (const int*)d_in[1];
    const int*   dst = (const int*)d_in[2];
    const float* W1  = (const float*)d_in[3];
    const float* b1  = (const float*)d_in[4];
    const float* W2  = (const float*)d_in[5];
    const float* b2  = (const float*)d_in[6];
    float* out = (float*)d_out;

    const long N = N_NODES;
    const int  E = N_EDGES;

    float* ws      = (float*)d_ws;
    float* outdeg  = ws;            // N (indeg directly after: zero 2N at once)
    float* indeg   = ws + N;
    float* outnorm = ws + 2 * N;
    float* innorm  = ws + 3 * N;
    float* h1      = ws + 4 * N;    // 64N
    float* h2      = ws + 4 * N;    // 32N (reuse of h1 region, after scatter1)
    float* agg2    = ws + 36 * N;   // 32N (reuse of h1 region, after scatter1)
    float* agg1    = ws + 68 * N;   // 64N; becomes y1 in-place
    float* y1      = agg1;
    v16bf* bfrag1  = (v16bf*)(ws + 132 * N);  // 16 frags * 32 lanes
    v16bf* bfrag2  = bfrag1 + 16 * 32;        // 4 frags * 32 lanes

    const int TPB = 256;

    // --- degrees + norms ---
    zero_f32<<<(int)((2 * N + TPB - 1) / TPB), TPB, 0, stream>>>(outdeg, 2 * N);
    zero_f32<<<(int)((64 * N + TPB - 1) / TPB), TPB, 0, stream>>>(agg1, 64 * N);
    degree_kernel<<<(E + TPB - 1) / TPB, TPB, 0, stream>>>(src, dst, outdeg, indeg, E);
    norm_kernel<<<(int)((N + TPB - 1) / TPB), TPB, 0, stream>>>(outdeg, indeg, outnorm, innorm, (int)N);

    // --- pre-pack weights into WMMA bf16 B-fragments ---
    prep_bfrag<4, 4><<<16, 32, 0, stream>>>(W1, bfrag1, 64);
    prep_bfrag<2, 2><<<4,  32, 0, stream>>>(W2, bfrag2, 32);

    // --- layer 1: GEMM (wmma bf16), gather/scatter, relu epilogue ---
    const int ntiles = N_NODES / 16;            // 6250
    const int gemm_blocks = (ntiles + 3) / 4;   // 4 waves (128 thr) per block
    gemm_wmma<4, 4><<<gemm_blocks, 128, 0, stream>>>(X, outnorm, bfrag1, h1);
    {
        long threads = (long)E * 16;            // F=64 -> 16 float4 chunks/edge
        scatter_add<64><<<(int)((threads + TPB - 1) / TPB), TPB, 0, stream>>>(h1, src, dst, agg1);
    }
    epilogue_kernel<64, true><<<(int)((64 * N + TPB - 1) / TPB), TPB, 0, stream>>>(agg1, innorm, b1, y1);

    // --- layer 2 (h1 region now free: holds h2 + agg2) ---
    zero_f32<<<(int)((32 * N + TPB - 1) / TPB), TPB, 0, stream>>>(agg2, 32 * N);
    gemm_wmma<2, 2><<<gemm_blocks, 128, 0, stream>>>(y1, outnorm, bfrag2, h2);
    {
        long threads = (long)E * 8;             // F=32 -> 8 float4 chunks/edge
        scatter_add<32><<<(int)((threads + TPB - 1) / TPB), TPB, 0, stream>>>(h2, src, dst, agg2);
    }
    epilogue_kernel<32, false><<<(int)((32 * N + TPB - 1) / TPB), TPB, 0, stream>>>(agg2, innorm, b2, out);
}
